// GraphNetworkBlock_63136019251925
// MI455X (gfx1250) — matrix operand order repaired and checked by reading, not compiled, not run on hardware
//
#include <hip/hip_runtime.h>

// ---------------------------------------------------------------------------
// GraphNetwork block for MI455X (gfx1250), bf16 WMMA + f32 accumulate.
// ---------------------------------------------------------------------------

#define N_NODES   50000
#define N_EDGES   800000
#define NODE_DIM  64
#define EDGE_DIM  64
#define HIDDEN    128
#define EDGE_IN   (EDGE_DIM + 2 * NODE_DIM)   // 192
#define NODE_IN   (NODE_DIM + EDGE_DIM)       // 128

typedef __attribute__((ext_vector_type(16))) __bf16 v16bf;
typedef __attribute__((ext_vector_type(8)))  __bf16 v8bf;
typedef __attribute__((ext_vector_type(8)))  float  v8f;
typedef __attribute__((ext_vector_type(4)))  float  v4f;

// Convert 8 contiguous f32 (32B-aligned) to 8 bf16.
__device__ inline v8bf cvt8(const float* __restrict__ p) {
    v4f a = *(const v4f*)(p);
    v4f b = *(const v4f*)(p + 4);
    v8bf r;
    r[0] = (__bf16)a[0]; r[1] = (__bf16)a[1]; r[2] = (__bf16)a[2]; r[3] = (__bf16)a[3];
    r[4] = (__bf16)b[0]; r[5] = (__bf16)b[1]; r[6] = (__bf16)b[2]; r[7] = (__bf16)b[3];
    return r;
}

__device__ inline v16bf cat16(v8bf lo, v8bf hi) {
    return __builtin_shufflevector(lo, hi, 0,1,2,3,4,5,6,7,8,9,10,11,12,13,14,15);
}

// B fragment: two 8-element bf16 runs from LDS-transposed weights (lane = column N,
// lanes 0-15 get K0+0..7,16..23, lanes 16-31 get K0+8..15,24..31 -> caller offset).
__device__ inline v16bf ldsB(const __bf16* __restrict__ w) {
    v8bf lo = *(const v8bf*)(w);
    v8bf hi = *(const v8bf*)(w + 16);
    return cat16(lo, hi);
}

__device__ inline v8f wmma_bf16(v16bf a, v16bf b, v8f c) {
    return __builtin_amdgcn_wmma_f32_16x16x32_bf16(
        /*neg_a=*/false, a, /*neg_b=*/false, b,
        /*c_mod=*/(short)0, c, /*reuse_a=*/false, /*reuse_b=*/false);
}

// ---------------------------------------------------------------------------
// Kernel 0: zero the message accumulator (deterministic per call).
// ---------------------------------------------------------------------------
__global__ __launch_bounds__(256) void gnb_zero_kernel(float* __restrict__ p, int n4) {
    int i = blockIdx.x * blockDim.x + threadIdx.x;
    if (i < n4) ((v4f*)p)[i] = (v4f){0.f, 0.f, 0.f, 0.f};
}

// ---------------------------------------------------------------------------
// Kernel 1: edge MLP + scatter-add of messages.
//   edge_in = [edge_attr | x[row] | x[col]]  (192)
//   h   = relu(edge_in @ eW1 + eb1)          (128)
//   out = h @ eW2 + eb2                      (64)  -> edge_out, atomic-add msg[col]
// One wave handles 16 edges per group; 8 waves per block.
// ---------------------------------------------------------------------------
__global__ __launch_bounds__(256) void gnb_edge_kernel(
    const float* __restrict__ x,
    const int*   __restrict__ rowIdx,
    const int*   __restrict__ colIdx,
    const float* __restrict__ edge_attr,
    const float* __restrict__ eW1, const float* __restrict__ eb1,
    const float* __restrict__ eW2, const float* __restrict__ eb2,
    float* __restrict__ edge_out,
    float* __restrict__ msg)
{
    __shared__ __bf16 Wt1[HIDDEN][EDGE_IN];    // 128 x 192 bf16, transposed eW1
    __shared__ __bf16 Wt2[EDGE_DIM][HIDDEN];   // 64 x 128 bf16, transposed eW2
    __shared__ __bf16 hbuf[8][16][HIDDEN];     // per-wave layer-1 output bounce

    // Stage weights transposed (Wt[n][k]) as bf16, coalesced global reads.
    for (int t = threadIdx.x; t < EDGE_IN * HIDDEN; t += 256) {
        int k = t / HIDDEN, n = t % HIDDEN;
        Wt1[n][k] = (__bf16)eW1[t];
    }
    for (int t = threadIdx.x; t < HIDDEN * EDGE_DIM; t += 256) {
        int k = t / EDGE_DIM, n = t % EDGE_DIM;
        Wt2[n][k] = (__bf16)eW2[t];
    }
    __syncthreads();

    const int lane  = threadIdx.x & 31;
    const int wave  = threadIdx.x >> 5;
    const int M     = lane & 15;               // A-matrix row handled by this lane
    const int koffA = (lane < 16) ? 0 : 8;     // A layout K offset
    const int koffB = (lane < 16) ? 0 : 8;     // B layout K offset
    const int bcol  = lane & 15;               // C/D column N
    const int hiM   = (lane < 16) ? 0 : 8;     // C/D row offset

    const int nGroups    = N_EDGES / 16;       // 50000, exact
    const int waveGlobal = blockIdx.x * 8 + wave;
    const int waveStride = gridDim.x * 8;

    for (int g = waveGlobal; g < nGroups; g += waveStride) {
        const int e0 = g * 16;
        const int e  = e0 + M;
        const float* s0 = edge_attr + e * EDGE_DIM;
        const float* s1 = x + rowIdx[e] * NODE_DIM;
        const float* s2 = x + colIdx[e] * NODE_DIM;

        // Build A fragments (16x192 in 6 K-steps of 32). Each lane needs two
        // contiguous 8-element runs per K-step; runs never cross a 64 boundary.
        v16bf A[6];
#pragma unroll
        for (int kb = 0; kb < 6; ++kb) {
            int sa = kb * 32 + koffA;
            int sb = sa + 16;
            const float* pa = (sa < 64) ? (s0 + sa) : ((sa < 128) ? (s1 + sa - 64) : (s2 + sa - 128));
            const float* pb = (sb < 64) ? (s0 + sb) : ((sb < 128) ? (s1 + sb - 64) : (s2 + sb - 128));
            A[kb] = cat16(cvt8(pa), cvt8(pb));
        }

        // Layer 1: h = relu(A @ W1 + b1), 8 N-tiles of 16.
#pragma unroll
        for (int nb = 0; nb < 8; ++nb) {
            v8f acc = {};
#pragma unroll
            for (int kb = 0; kb < 6; ++kb) {
                v16bf b = ldsB(&Wt1[nb * 16 + bcol][kb * 32 + koffB]);
                acc = wmma_bf16(A[kb], b, acc);
            }
            float bias = eb1[nb * 16 + bcol];
#pragma unroll
            for (int r = 0; r < 8; ++r) {
                float v = acc[r] + bias;
                v = v > 0.f ? v : 0.f;
                hbuf[wave][r + hiM][nb * 16 + bcol] = (__bf16)v;
            }
        }
        // Cross-lane LDS RAW within the wave: pin ordering + wait.
        asm volatile("s_wait_dscnt 0" ::: "memory");

        // Layer 2 A fragments from bounced h (16x128, 4 K-steps).
        v16bf A2[4];
#pragma unroll
        for (int kb = 0; kb < 4; ++kb) {
            int sa = kb * 32 + koffA;
            v8bf lo = *(const v8bf*)&hbuf[wave][M][sa];
            v8bf hi = *(const v8bf*)&hbuf[wave][M][sa + 16];
            A2[kb] = cat16(lo, hi);
        }
        asm volatile("" ::: "memory");  // keep next-iter hbuf writes below these reads

        // Layer 2: out = h @ W2 + b2, 4 N-tiles; store + scatter-add messages.
#pragma unroll
        for (int nb = 0; nb < 4; ++nb) {
            v8f acc = {};
#pragma unroll
            for (int kb = 0; kb < 4; ++kb) {
                v16bf b = ldsB(&Wt2[nb * 16 + bcol][kb * 32 + koffB]);
                acc = wmma_bf16(A2[kb], b, acc);
            }
            float bias = eb2[nb * 16 + bcol];
#pragma unroll
            for (int r = 0; r < 8; ++r) {
                float v = acc[r] + bias;
                int er = e0 + r + hiM;
                edge_out[er * EDGE_DIM + nb * 16 + bcol] = v;
                int c = colIdx[er];
                (void)__hip_atomic_fetch_add(&msg[c * EDGE_DIM + nb * 16 + bcol], v,
                                             __ATOMIC_RELAXED, __HIP_MEMORY_SCOPE_AGENT);
            }
        }
    }
}

// ---------------------------------------------------------------------------
// Kernel 2: node MLP.
//   node_in = [x | messages] (128); h2 = relu(node_in @ nW1 + nb1);
//   x_new = h2 @ nW2 + nb2 (64)
// ---------------------------------------------------------------------------
__global__ __launch_bounds__(256) void gnb_node_kernel(
    const float* __restrict__ x,
    const float* __restrict__ msg,
    const float* __restrict__ nW1, const float* __restrict__ nb1,
    const float* __restrict__ nW2, const float* __restrict__ nb2,
    float* __restrict__ x_new)
{
    __shared__ __bf16 Wt1[HIDDEN][NODE_IN];    // 128 x 128 bf16
    __shared__ __bf16 Wt2[NODE_DIM][HIDDEN];   // 64 x 128 bf16
    __shared__ __bf16 hbuf[8][16][HIDDEN];

    for (int t = threadIdx.x; t < NODE_IN * HIDDEN; t += 256) {
        int k = t / HIDDEN, n = t % HIDDEN;
        Wt1[n][k] = (__bf16)nW1[t];
    }
    for (int t = threadIdx.x; t < HIDDEN * NODE_DIM; t += 256) {
        int k = t / NODE_DIM, n = t % NODE_DIM;
        Wt2[n][k] = (__bf16)nW2[t];
    }
    __syncthreads();

    const int lane  = threadIdx.x & 31;
    const int wave  = threadIdx.x >> 5;
    const int M     = lane & 15;
    const int koffA = (lane < 16) ? 0 : 8;
    const int koffB = (lane < 16) ? 0 : 8;
    const int bcol  = lane & 15;
    const int hiM   = (lane < 16) ? 0 : 8;

    const int nGroups    = N_NODES / 16;       // 3125, exact
    const int waveGlobal = blockIdx.x * 8 + wave;
    const int waveStride = gridDim.x * 8;

    for (int g = waveGlobal; g < nGroups; g += waveStride) {
        const int n0 = g * 16;
        const int nd = n0 + M;
        const float* s0 = x   + nd * NODE_DIM;
        const float* s1 = msg + nd * EDGE_DIM;

        v16bf A[4];
#pragma unroll
        for (int kb = 0; kb < 4; ++kb) {
            int sa = kb * 32 + koffA;
            int sb = sa + 16;
            const float* pa = (sa < 64) ? (s0 + sa) : (s1 + sa - 64);
            const float* pb = (sb < 64) ? (s0 + sb) : (s1 + sb - 64);
            A[kb] = cat16(cvt8(pa), cvt8(pb));
        }

#pragma unroll
        for (int nb = 0; nb < 8; ++nb) {
            v8f acc = {};
#pragma unroll
            for (int kb = 0; kb < 4; ++kb) {
                v16bf b = ldsB(&Wt1[nb * 16 + bcol][kb * 32 + koffB]);
                acc = wmma_bf16(A[kb], b, acc);
            }
            float bias = nb1[nb * 16 + bcol];
#pragma unroll
            for (int r = 0; r < 8; ++r) {
                float v = acc[r] + bias;
                v = v > 0.f ? v : 0.f;
                hbuf[wave][r + hiM][nb * 16 + bcol] = (__bf16)v;
            }
        }
        asm volatile("s_wait_dscnt 0" ::: "memory");

        v16bf A2[4];
#pragma unroll
        for (int kb = 0; kb < 4; ++kb) {
            int sa = kb * 32 + koffA;
            v8bf lo = *(const v8bf*)&hbuf[wave][M][sa];
            v8bf hi = *(const v8bf*)&hbuf[wave][M][sa + 16];
            A2[kb] = cat16(lo, hi);
        }
        asm volatile("" ::: "memory");

#pragma unroll
        for (int nb = 0; nb < 4; ++nb) {
            v8f acc = {};
#pragma unroll
            for (int kb = 0; kb < 4; ++kb) {
                v16bf b = ldsB(&Wt2[nb * 16 + bcol][kb * 32 + koffB]);
                acc = wmma_bf16(A2[kb], b, acc);
            }
            float bias = nb2[nb * 16 + bcol];
#pragma unroll
            for (int r = 0; r < 8; ++r) {
                float v = acc[r] + bias;
                int nr = n0 + r + hiM;
                x_new[nr * NODE_DIM + nb * 16 + bcol] = v;
            }
        }
    }
}

// ---------------------------------------------------------------------------
extern "C" void kernel_launch(void* const* d_in, const int* in_sizes, int n_in,
                              void* d_out, int out_size, void* d_ws, size_t ws_size,
                              hipStream_t stream) {
    (void)in_sizes; (void)n_in; (void)out_size; (void)ws_size;
    const float* x         = (const float*)d_in[0];
    const int*   edge_idx  = (const int*)  d_in[1];   // [2, E]
    const float* edge_attr = (const float*)d_in[2];
    const float* eW1       = (const float*)d_in[3];
    const float* eb1       = (const float*)d_in[4];
    const float* eW2       = (const float*)d_in[5];
    const float* eb2       = (const float*)d_in[6];
    const float* nW1       = (const float*)d_in[7];
    const float* nb1       = (const float*)d_in[8];
    const float* nW2       = (const float*)d_in[9];
    const float* nb2       = (const float*)d_in[10];

    float* x_new    = (float*)d_out;                              // [N, 64]
    float* edge_out = (float*)d_out + (size_t)N_NODES * NODE_DIM; // [E, 64]
    float* msg      = (float*)d_ws;                               // [N, 64] accumulator

    const int zn4 = (N_NODES * NODE_DIM) / 4;
    gnb_zero_kernel<<<(zn4 + 255) / 256, 256, 0, stream>>>(msg, zn4);

    gnb_edge_kernel<<<1563, 256, 0, stream>>>(
        x, edge_idx, edge_idx + N_EDGES, edge_attr,
        eW1, eb1, eW2, eb2, edge_out, msg);

    gnb_node_kernel<<<391, 256, 0, stream>>>(
        x, msg, nW1, nb1, nW2, nb2, x_new);
}
